// RNNCell_29360396435533
// MI455X (gfx1250) — compile-verified, matrix operand order checked by reading, not run on hardware
//
#include <hip/hip_runtime.h>
#include <hip/hip_bf16.h>
#include <math.h>

typedef __attribute__((ext_vector_type(16))) __bf16 v16bf;
typedef __attribute__((ext_vector_type(8)))  float  v8f;

#define N_B 32
#define L_S 2048
#define H_D 512
#define H2  1024

// ---- workspace layout (bytes) ----
#define OFF_XPROJ ((size_t)0)
#define SZ_XPROJ  ((size_t)N_B * L_S * H_D * 4)   // 128 MB fp32 x_proj
#define OFF_WBX   (OFF_XPROJ + SZ_XPROJ)
#define SZ_WB     ((size_t)H_D * H_D * 2)         // 512 KB bf16 repacked weights
#define OFF_WBH   (OFF_WBX + SZ_WB)
#define OFF_H0    (OFF_WBH + SZ_WB)
#define SZ_H      ((size_t)N_B * H_D * 2)         // 32 KB bf16 hidden state
#define OFF_H1    (OFF_H0 + SZ_H)
#define OFF_CTR   (OFF_H1 + SZ_H)

__device__ __forceinline__ float fast_tanh(float x) {
#if __has_builtin(__builtin_amdgcn_tanhf)
  return __builtin_amdgcn_tanhf(x);   // v_tanh_f32 on gfx1250
#else
  return tanhf(x);
#endif
}

// Repack W (fp32, [o][2H]) into bf16 B-operand order:
//   WB[(kt*2+kh)*512 + n][j]  holds  B[k=kt*32+kh*16+j, n]
// so lane l of a wave (n = l%16 + 16*nt, kh = l/16) reads one contiguous
// 32-byte v16bf fragment per K-tile. Also zero h0 and the barrier counter.
__global__ void prep_kernel(const float* __restrict__ W, char* __restrict__ ws) {
  __bf16* wbx = (__bf16*)(ws + OFF_WBX);
  __bf16* wbh = (__bf16*)(ws + OFF_WBH);
  __bf16* h0  = (__bf16*)(ws + OFF_H0);
  unsigned int* ctr = (unsigned int*)(ws + OFF_CTR);

  int i = blockIdx.x * blockDim.x + threadIdx.x;       // 0 .. 512*512-1
  if (i < H_D * H_D) {
    int j  = i & 15;
    int n  = (i >> 4) & (H_D - 1);
    int g  = i >> 13;                                  // kt*2 + kh
    int kh = g & 1;
    int kt = g >> 1;
    int k  = kt * 32 + kh * 16 + j;
    wbx[i] = (__bf16)W[(size_t)n * H2 + H_D + k];      // W_x[o=n, k]
    wbh[i] = (__bf16)W[(size_t)n * H2 + k];            // W_h[o=n, k]
  }
  if (i < N_B * H_D) h0[i] = (__bf16)0.0f;
  if (i == 0) *ctr = 0u;
}

// x_proj = x @ W_x^T + b.   One wave per 16x16 output tile.
// M = N*L = 65536 rows (4096 tiles), N = 512 cols (32 tiles).
__global__ void __launch_bounds__(256) xproj_kernel(const float* __restrict__ x,
                                                    const float* __restrict__ b,
                                                    char* __restrict__ ws) {
  float*        xproj = (float*)(ws + OFF_XPROJ);
  const __bf16* wbx   = (const __bf16*)(ws + OFF_WBX);

  int tid  = blockIdx.x * blockDim.x + threadIdx.x;
  int wave = tid >> 5;
  int lane = threadIdx.x & 31;
  int mt = wave >> 5;                 // 0..4095 row tile over N*L
  int nt = wave & 31;                 // 0..31   col tile over H
  int rowbase = mt * 16;
  int kh  = lane >> 4;
  int col = nt * 16 + (lane & 15);

  const float* arow = x + (size_t)(rowbase + (lane & 15)) * H_D + kh * 16;

  v8f c = {};
#pragma unroll
  for (int kt = 0; kt < 16; ++kt) {
    v16bf a;
    const float* ap = arow + kt * 32;
#pragma unroll
    for (int j = 0; j < 16; ++j) a[j] = (__bf16)ap[j];
    v16bf bm = *(const v16bf*)(wbx + ((size_t)(kt * 2 + kh) * H_D + col) * 16);
    c = __builtin_amdgcn_wmma_f32_16x16x32_bf16(false, a, false, bm,
                                                (short)0, c, false, false);
  }
  float bias = b[col];
#pragma unroll
  for (int r = 0; r < 8; ++r) {
    int row = rowbase + r + 8 * kh;   // C layout: VGPR r, lane-half selects M+8
    xproj[(size_t)row * H_D + col] = c[r] + bias;
  }
}

// Persistent recurrence kernel: 8 blocks x 8 waves = 64 waves,
// one 16x16 tile of h_new (32 x 512) per wave. W_h slice lives in
// 128 VGPRs per wave for all 2048 steps. Each step:
//   1) async-copy the block's 16 rows of h_prev (16 KB bf16) into LDS
//      with global_load_async_to_lds_b128 (ASYNCcnt path)
//   2) WMMA A operands come from LDS (ds_load), B from registers,
//      C initialized from x_proj
//   3) tanh -> fp32 output + bf16 next state to the global ping-pong
//   4) cumulative-atomic grid barrier
__global__ void __launch_bounds__(256, 1) rnn_kernel(float* __restrict__ out,
                                                     char* __restrict__ ws) {
  const float*  xproj = (const float*)(ws + OFF_XPROJ);
  const __bf16* wbh   = (const __bf16*)(ws + OFF_WBH);
  __bf16* h0 = (__bf16*)(ws + OFF_H0);
  __bf16* h1 = (__bf16*)(ws + OFF_H1);
  unsigned int* ctr = (unsigned int*)(ws + OFF_CTR);

  __shared__ __bf16 smem[16 * H_D];   // 16 KB: this block's 16 rows of h_prev

  int wave = (blockIdx.x * blockDim.x + threadIdx.x) >> 5;  // 0..63
  int lane = threadIdx.x & 31;
  int mt = wave >> 5;                 // 0..1  (batch tile; uniform per block)
  int nt = wave & 31;                 // 0..31 (output-col tile)
  int kh  = lane >> 4;
  int col = nt * 16 + (lane & 15);
  int lrow = lane & 15;               // local A row inside LDS tile

  // Resident B operand: W_h^T K-slice for our 16 columns (512x16 bf16).
  v16bf wb[16];
#pragma unroll
  for (int kt = 0; kt < 16; ++kt)
    wb[kt] = *(const v16bf*)(wbh + ((size_t)(kt * 2 + kh) * H_D + col) * 16);

  const unsigned int nblocks = gridDim.x;
  const unsigned int lds_base = (unsigned int)(size_t)(&smem[0]); // LDS offset = addr[31:0]

  for (int t = 0; t < L_S; ++t) {
    const __bf16* hprev = (t & 1) ? h1 : h0;
    __bf16*       hnext = (t & 1) ? h0 : h1;

    // ---- stage h_prev rows [mt*16 .. mt*16+15] into LDS (async path) ----
    {
      unsigned long long gbase =
          (unsigned long long)(size_t)(hprev + (size_t)mt * 16 * H_D);
#pragma unroll
      for (int r = 0; r < 4; ++r) {
        unsigned int off = ((unsigned int)threadIdx.x + r * 256u) * 16u; // 16 B/lane
        unsigned int laddr = lds_base + off;
        unsigned long long gaddr = gbase + off;
        asm volatile("global_load_async_to_lds_b128 %0, %1, off"
                     :: "v"(laddr), "v"(gaddr) : "memory");
      }
      asm volatile("s_wait_asynccnt 0" ::: "memory");
      __syncthreads();
    }

    // ---- C init = x_proj tile at timestep t ----
    v8f c;
#pragma unroll
    for (int r = 0; r < 8; ++r) {
      int m = mt * 16 + r + 8 * kh;
      c[r] = xproj[((size_t)m * L_S + t) * H_D + col];
    }

    // ---- 16 WMMAs over K = 512, A from LDS, B from registers ----
#pragma unroll
    for (int kt = 0; kt < 16; ++kt) {
      v16bf a = *(const v16bf*)(&smem[lrow * H_D + kt * 32 + kh * 16]);
      c = __builtin_amdgcn_wmma_f32_16x16x32_bf16(false, a, false, wb[kt],
                                                  (short)0, c, false, false);
    }

    // ---- tanh, emit fp32 output and bf16 next-state ----
#pragma unroll
    for (int r = 0; r < 8; ++r) {
      int m = mt * 16 + r + 8 * kh;
      float v = fast_tanh(c[r]);
      out[(size_t)m * L_S * H_D + (size_t)t * H_D + col] = v;
      hnext[m * H_D + col] = (__bf16)v;
    }

    // ---- grid-wide barrier (also protects LDS reuse across steps) ----
    __syncthreads();
    if (threadIdx.x == 0) {
      __hip_atomic_fetch_add(ctr, 1u, __ATOMIC_ACQ_REL, __HIP_MEMORY_SCOPE_AGENT);
      unsigned int target = nblocks * (unsigned int)(t + 1);
      while (__hip_atomic_load(ctr, __ATOMIC_ACQUIRE, __HIP_MEMORY_SCOPE_AGENT) < target) {
        __builtin_amdgcn_s_sleep(1);
      }
    }
    __syncthreads();
  }
}

extern "C" void kernel_launch(void* const* d_in, const int* in_sizes, int n_in,
                              void* d_out, int out_size, void* d_ws, size_t ws_size,
                              hipStream_t stream) {
  const float* x = (const float*)d_in[0];   // (32, 2048, 512) f32
  const float* W = (const float*)d_in[1];   // (512, 1024) f32
  const float* b = (const float*)d_in[2];   // (512,) f32
  float* out = (float*)d_out;               // (32, 2048, 512) f32
  char* ws = (char*)d_ws;

  (void)in_sizes; (void)n_in; (void)out_size; (void)ws_size;

  // 1) repack weights to bf16 WMMA layout, zero h0 + barrier counter
  prep_kernel<<<(H_D * H_D + 255) / 256, 256, 0, stream>>>(W, ws);
  // 2) x_proj = x @ W_x^T + b   (65536 x 512 x 512, 131072 waves)
  xproj_kernel<<<16384, 256, 0, stream>>>(x, b, ws);
  // 3) sequential recurrence, persistent 8-block kernel, async-LDS staging
  rnn_kernel<<<8, 256, 0, stream>>>(out, ws);
}
